// TransformerEncoder_47141561041530
// MI455X (gfx1250) — matrix operand order repaired and checked by reading, not compiled, not run on hardware
//
#include <hip/hip_runtime.h>
#include <hip/hip_bf16.h>

// MI455X (gfx1250) transformer encoder forward.
// bf16 data / fp32-accumulate WMMA (v_wmma_f32_16x16x32_bf16) for all GEMMs +
// flash attention; fp32 epilogues. GEMM B-panels staged in LDS via the Tensor
// Data Mover (tensor_load_to_lds + s_wait_tensorcnt) with double buffering.

typedef __bf16 bf16;
typedef __attribute__((ext_vector_type(8)))  __bf16 v8bf;
typedef __attribute__((ext_vector_type(16))) __bf16 v16bf;
typedef __attribute__((ext_vector_type(8)))  float  v8f;
typedef __attribute__((ext_vector_type(4)))  int    v4i;
typedef __attribute__((ext_vector_type(8)))  int    v8i;
typedef __attribute__((ext_vector_type(4)))  unsigned int v4u;

#if defined(__HIP_DEVICE_COMPILE__) && __has_builtin(__builtin_amdgcn_tensor_load_to_lds)
#define HAVE_TDM 1
#if __has_include(<hip/amd_detail/amd_gfx1250_TDM.h>)
#define TDM_6ARG 1   // therock/clang-23 toolchain: 6-arg builtin
#else
#define TDM_6ARG 0   // ROCm 7.2 / clang-22: 5-arg builtin
#endif
#else
#define HAVE_TDM 0
#endif

static __device__ __forceinline__ v16bf frag2(const bf16* p0, const bf16* p1) {
  v8bf a = *(const v8bf*)p0;
  v8bf b = *(const v8bf*)p1;
  v16bf r;
#pragma unroll
  for (int i = 0; i < 8; ++i) { r[i] = a[i]; r[i + 8] = b[i]; }
  return r;
}

static __device__ __forceinline__ v8f wmma_bf16(v16bf a, v16bf b, v8f c) {
  return __builtin_amdgcn_wmma_f32_16x16x32_bf16(
      /*neg_a=*/false, a, /*neg_b=*/false, b,
      /*c_mod=*/(short)0, c, /*reuse_a=*/false, /*reuse_b=*/false);
}

// ---------------------------------------------------------------------------
// Embedding + sinusoidal positional encoding -> x (fp32) and xb (bf16)
// ---------------------------------------------------------------------------
__global__ __launch_bounds__(256) void embed_pe(
    const int* __restrict__ tok, const float* __restrict__ emb,
    float* __restrict__ x, bf16* __restrict__ xb) {
  const int D = 1024, T = 2048;
  int row = blockIdx.x;           // b*T + t
  int tid = threadIdx.x;
  int t = row & (T - 1);
  int tk = tok[row];
#pragma unroll
  for (int i = 0; i < 4; ++i) {
    int c = tid + i * 256;
    float div = __expf(-(float)(c & ~1) * (9.2103403719761836f / 1024.0f));
    float ang = (float)t * div;
    float pe = (c & 1) ? __cosf(ang) : __sinf(ang);
    float v = emb[(size_t)tk * D + c] + pe;
    x[(size_t)row * D + c] = v;
    xb[(size_t)row * D + c] = (bf16)v;
  }
}

// ---------------------------------------------------------------------------
// Weight convert + transpose: Wt[n*K + k] = (bf16) W[k*N + n]
// ---------------------------------------------------------------------------
__global__ __launch_bounds__(256) void wtranspose(
    const float* __restrict__ W, bf16* __restrict__ Wt, int K, int N) {
  size_t i = (size_t)blockIdx.x * 256 + threadIdx.x;
  size_t total = (size_t)K * N;
  if (i >= total) return;
  size_t k = i % (size_t)K;
  size_t n = i / (size_t)K;
  Wt[i] = (bf16)W[k * (size_t)N + n];
}

// ---------------------------------------------------------------------------
// Repack V out of qkv into V^T layout: Vt[((b*16+h)*64+d)*2048 + t]
// ---------------------------------------------------------------------------
__global__ __launch_bounds__(256) void repack_vt(
    const bf16* __restrict__ qkv, bf16* __restrict__ Vt) {
  size_t i = (size_t)blockIdx.x * 256 + threadIdx.x;   // over 8192*1024
  int t = (int)(i & 2047);
  size_t bhd = i >> 11;
  int d = (int)(bhd & 63);
  size_t bh = bhd >> 6;
  int h = (int)(bh & 15);
  int b = (int)(bh >> 4);
  Vt[i] = qkv[((size_t)(b * 2048 + t)) * 3072 + 2048 + h * 64 + d];
}

// ---------------------------------------------------------------------------
// WMMA GEMM: C[M,N] = A[M,K](bf16 row-major) * Wt[N,K]^T + bias, opt. relu.
// Block = 128 rows x 64 cols: 8 waves x (16 rows x 4 n-tiles). The 64x32 B
// panel (shared by all 8 waves) is staged in LDS, double-buffered, filled by
// the TDM (wave 0 issues tensor_load_to_lds, overlapped with WMMA on the
// other buffer) or by a cooperative copy when the builtin is unavailable.
// LDS rows padded 64B + 16B to spread b128 reads across banks (TDM pad
// fields: interval = 16 DWORDs, amount = 4 DWORDs).
// ---------------------------------------------------------------------------
__global__ __launch_bounds__(256) void gemm_bf16_wmma(
    const bf16* __restrict__ A, const bf16* __restrict__ Wt,
    const float* __restrict__ bias, float* __restrict__ Cf,
    bf16* __restrict__ Cb, int M, int N, int K, int relu) {
  __shared__ __align__(16) bf16 ldsB[2][64 * 40];   // 2 x 5 KB, padded rows
  int lane = threadIdx.x & 31;
  int w    = threadIdx.x >> 5;
  int mtile = blockIdx.x * 8 + w;
  int nbase = blockIdx.y * 64;
  int m  = lane & 15;
  int hi = lane >> 4;

  const bf16* ap = A + (size_t)(mtile * 16 + m) * K + hi * 16;

  auto stageB = [&](int buf, int k0) {
#if HAVE_TDM
    if (w == 0) {
      unsigned int ldsoff = (unsigned int)(size_t)&ldsB[buf][0];
      unsigned long long ga =
          (unsigned long long)(size_t)(Wt + (size_t)nbase * K + k0);
      // D# group0: count=1 | lds_addr | global_addr[56:0] | type=2
      v4u g0 = { 1u, ldsoff, (unsigned int)ga,
                 (unsigned int)((ga >> 32) & 0x01FFFFFFu) | 0x80000000u };
      // D# group1: data_size=2B, pad_enable, pad_interval=16DW, pad_amount=4DW,
      // tensor_dim0=K, tensor_dim1=N, tile_dim0=32, tile_dim1=64, stride0=K.
      v8i g1 = { (int)0x06D10000u,
                 (int)(((unsigned)K & 0xFFFFu) << 16),
                 (int)((((unsigned)K >> 16) & 0xFFFFu) |
                       (((unsigned)N & 0xFFFFu) << 16)),
                 (int)((((unsigned)N >> 16) & 0xFFFFu) | (32u << 16)),
                 64,
                 (int)(unsigned)K,
                 0, 0 };
      v4i z4 = {};
#if TDM_6ARG
      v8i z8 = {};
      __builtin_amdgcn_tensor_load_to_lds(g0, g1, z4, z4, z8, 0);
#else
      __builtin_amdgcn_tensor_load_to_lds(g0, g1, z4, z4, 0);
#endif
    }
#else
    int n_ = (int)threadIdx.x >> 2;          // 0..63 panel rows
    int kc = ((int)threadIdx.x & 3) * 8;     // 0/8/16/24 halves
    *(v8bf*)&ldsB[buf][n_ * 40 + kc] =
        *(const v8bf*)&Wt[(size_t)(nbase + n_) * K + k0 + kc];
#endif
  };

  v8f acc[4] = {};
  stageB(0, 0);
#if HAVE_TDM
  if (w == 0) __builtin_amdgcn_s_wait_tensorcnt(0);
#endif
  __syncthreads();

  for (int k0 = 0; k0 < K; k0 += 32) {
    int cur = (k0 >> 5) & 1;
    if (k0 + 32 < K) stageB(1 - cur, k0 + 32);   // async fill of next buffer
    __builtin_prefetch(ap + k0 + 256, 0, 1);
    v16bf af = frag2(ap + k0, ap + k0 + 8);
    const bf16* lb = &ldsB[cur][0] + m * 40 + hi * 16;
#pragma unroll
    for (int t = 0; t < 4; ++t) {
      v16bf bfr = frag2(lb + t * 16 * 40, lb + t * 16 * 40 + 8);
      acc[t] = wmma_bf16(af, bfr, acc[t]);
    }
#if HAVE_TDM
    if (w == 0) __builtin_amdgcn_s_wait_tensorcnt(0);  // after compute: overlap
#endif
    __syncthreads();
  }

#pragma unroll
  for (int t = 0; t < 4; ++t) {
    int ncol = nbase + t * 16 + m;
    float bv = bias ? bias[ncol] : 0.0f;
#pragma unroll
    for (int r = 0; r < 8; ++r) {
      float v = acc[t][r] + bv;
      if (relu) v = fmaxf(v, 0.0f);
      size_t idx = (size_t)(mtile * 16 + hi * 8 + r) * N + ncol;
      if (Cf) Cf[idx] = v;
      if (Cb) Cb[idx] = (bf16)v;
    }
  }
}

// ---------------------------------------------------------------------------
// Flash attention: one wave per 16-query tile of one (b,h).
// qkv bf16 [B,T,3D] supplies Q and K rows directly; Vt is [B,H,dk,T].
// Online softmax in fp32; P converted C-layout -> A-layout through LDS.
// ---------------------------------------------------------------------------
__global__ __launch_bounds__(256) void flash_attn(
    const bf16* __restrict__ qkv, const bf16* __restrict__ Vt,
    bf16* __restrict__ Obf) {
  const int T = 2048, D3 = 3072, D = 1024, dk = 64;
  __shared__ __align__(16) bf16 plds[8][16 * 32];

  int lane = threadIdx.x & 31;
  int w    = threadIdx.x >> 5;
  int g  = blockIdx.x * 8 + w;       // global wave id
  int qt = g & 127;                  // T/16 = 128 query tiles
  int h  = (g >> 7) & 15;
  int b  = g >> 11;
  int m  = lane & 15;
  int hi = lane >> 4;
  int kbA = hi * 16;                 // A-frag K base

  const bf16* qrow = qkv + (size_t)(b * T + qt * 16 + m) * D3 + h * dk;
  v16bf qf0 = frag2(qrow + kbA,      qrow + kbA + 8);
  v16bf qf1 = frag2(qrow + 32 + kbA, qrow + 32 + kbA + 8);

  v8f oacc[4] = {};
  float mrun[8], lrun[8];
#pragma unroll
  for (int r = 0; r < 8; ++r) { mrun[r] = -1e30f; lrun[r] = 0.0f; }

  const bf16* kbasep = qkv + (size_t)b * T * D3 + D + h * dk;   // + key*D3
  const bf16* vbase  = Vt + ((size_t)(b * 16 + h)) * dk * T;    // + d*T + t
  bf16* pw = &plds[w][0];

  for (int kb = 0; kb < T; kb += 32) {
    v8f s0 = {}, s1 = {};
    {
      const bf16* kr0 = kbasep + (size_t)(kb + m) * D3 + hi * 16;
      const bf16* kr1 = kbasep + (size_t)(kb + 16 + m) * D3 + hi * 16;
      v16bf b00 = frag2(kr0,      kr0 + 8);
      v16bf b01 = frag2(kr0 + 32, kr0 + 40);
      v16bf b10 = frag2(kr1,      kr1 + 8);
      v16bf b11 = frag2(kr1 + 32, kr1 + 40);
      s0 = wmma_bf16(qf0, b00, s0);
      s0 = wmma_bf16(qf1, b01, s0);
      s1 = wmma_bf16(qf0, b10, s1);
      s1 = wmma_bf16(qf1, b11, s1);
    }
#pragma unroll
    for (int r = 0; r < 8; ++r) {
      float a0 = s0[r] * 0.125f;   // 1/sqrt(64)
      float a1 = s1[r] * 0.125f;
      float mx = fmaxf(a0, a1);
#pragma unroll
      for (int off = 8; off >= 1; off >>= 1)
        mx = fmaxf(mx, __shfl_xor(mx, off, 32));
      float mnew = fmaxf(mrun[r], mx);
      float p0 = __expf(a0 - mnew);
      float p1 = __expf(a1 - mnew);
      float rs = p0 + p1;
#pragma unroll
      for (int off = 8; off >= 1; off >>= 1)
        rs += __shfl_xor(rs, off, 32);
      float corr = __expf(mrun[r] - mnew);
      lrun[r] = lrun[r] * corr + rs;
      mrun[r] = mnew;
      oacc[0][r] *= corr; oacc[1][r] *= corr;
      oacc[2][r] *= corr; oacc[3][r] *= corr;
      int mm = r + hi * 8;
      pw[mm * 32 + m]      = (bf16)p0;
      pw[mm * 32 + 16 + m] = (bf16)p1;
    }
    __syncthreads();   // LDS P writes -> A-layout reads (uniform trip count)
    const bf16* pr = pw + m * 32 + kbA;
    v16bf pf = frag2(pr, pr + 8);
#pragma unroll
    for (int nd = 0; nd < 4; ++nd) {
      const bf16* vp = vbase + (size_t)(nd * 16 + m) * T + kb + hi * 16;
      v16bf vf = frag2(vp, vp + 8);
      oacc[nd] = wmma_bf16(pf, vf, oacc[nd]);
    }
  }

#pragma unroll
  for (int nd = 0; nd < 4; ++nd) {
#pragma unroll
    for (int r = 0; r < 8; ++r) {
      int t = qt * 16 + hi * 8 + r;
      int col = h * dk + nd * 16 + m;
      Obf[(size_t)(b * T + t) * D + col] = (bf16)(oacc[nd][r] / lrun[r]);
    }
  }
}

// ---------------------------------------------------------------------------
// x = LayerNorm(x + y) * g + be ; also writes bf16 copy xb. One block per row.
// ---------------------------------------------------------------------------
__global__ __launch_bounds__(256) void residual_ln(
    float* __restrict__ x, const float* __restrict__ y,
    const float* __restrict__ g, const float* __restrict__ be,
    bf16* __restrict__ xb) {
  const int D = 1024;
  __shared__ float red[256];
  int row = blockIdx.x;
  int tid = threadIdx.x;
  float v[4];
  float s = 0.0f;
#pragma unroll
  for (int i = 0; i < 4; ++i) {
    int c = tid + i * 256;
    v[i] = x[(size_t)row * D + c] + y[(size_t)row * D + c];
    s += v[i];
  }
  red[tid] = s;
  __syncthreads();
  for (int st = 128; st > 0; st >>= 1) {
    if (tid < st) red[tid] += red[tid + st];
    __syncthreads();
  }
  float mu = red[0] * (1.0f / D);
  __syncthreads();
  float s2 = 0.0f;
#pragma unroll
  for (int i = 0; i < 4; ++i) { float d = v[i] - mu; s2 += d * d; }
  red[tid] = s2;
  __syncthreads();
  for (int st = 128; st > 0; st >>= 1) {
    if (tid < st) red[tid] += red[tid + st];
    __syncthreads();
  }
  float inv = rsqrtf(red[0] * (1.0f / D) + 1e-5f);
#pragma unroll
  for (int i = 0; i < 4; ++i) {
    int c = tid + i * 256;
    float o = (v[i] - mu) * inv * g[c] + be[c];
    x[(size_t)row * D + c] = o;
    xb[(size_t)row * D + c] = (bf16)o;
  }
}

// ---------------------------------------------------------------------------
extern "C" void kernel_launch(void* const* d_in, const int* in_sizes, int n_in,
                              void* d_out, int out_size, void* d_ws, size_t ws_size,
                              hipStream_t stream) {
  (void)in_sizes; (void)n_in; (void)out_size; (void)ws_size;
  const int*   tokens = (const int*)  d_in[0];
  const float* emb    = (const float*)d_in[1];
  const float* Wqkv   = (const float*)d_in[2];
  const float* bqkv   = (const float*)d_in[3];
  const float* Wo     = (const float*)d_in[4];
  const float* bo     = (const float*)d_in[5];
  const float* g1     = (const float*)d_in[6];
  const float* be1    = (const float*)d_in[7];
  const float* W1     = (const float*)d_in[8];
  const float* bf1    = (const float*)d_in[9];
  const float* W2     = (const float*)d_in[10];
  const float* bf2    = (const float*)d_in[11];
  const float* g2     = (const float*)d_in[12];
  const float* be2    = (const float*)d_in[13];
  float* x = (float*)d_out;   // running fp32 activation; final LN lands here

  const int B = 4, T = 2048, D = 1024, DFF = 2048, H = 16, L = 6;
  const int M = B * T;   // 8192

  char* p = (char*)d_ws;
  auto carve = [&](size_t bytes) -> char* {
    char* r = p;
    p += (bytes + 255) & ~(size_t)255;
    return r;
  };
  bf16*  xb   = (bf16*) carve((size_t)M * D * 2);        // bf16 activations
  bf16*  qkv  = (bf16*) carve((size_t)M * 3 * D * 2);    // QKV bf16
  bf16*  Vt   = (bf16*) carve((size_t)M * D * 2);        // V^T [B,H,dk,T]
  bf16*  obf  = (bf16*) carve((size_t)M * D * 2);        // attention output
  bf16*  hbf  = (bf16*) carve((size_t)M * DFF * 2);      // FFN hidden
  float* tmp  = (float*)carve((size_t)M * D * 4);        // pre-LN fp32
  bf16*  Wt   = (bf16*) carve((size_t)3 * D * D * 2);    // transposed weight

  embed_pe<<<M, 256, 0, stream>>>(tokens, emb, x, xb);

  for (int l = 0; l < L; ++l) {
    // --- attention block ---
    wtranspose<<<(3 * D * D + 255) / 256, 256, 0, stream>>>(
        Wqkv + (size_t)l * D * 3 * D, Wt, D, 3 * D);
    gemm_bf16_wmma<<<dim3(M / 128, 3 * D / 64), 256, 0, stream>>>(
        xb, Wt, bqkv + (size_t)l * 3 * D, nullptr, qkv, M, 3 * D, D, 0);
    repack_vt<<<(M * D) / 256, 256, 0, stream>>>(qkv, Vt);
    flash_attn<<<(M / 16) * H / 8, 256, 0, stream>>>(qkv, Vt, obf);
    wtranspose<<<(D * D + 255) / 256, 256, 0, stream>>>(
        Wo + (size_t)l * D * D, Wt, D, D);
    gemm_bf16_wmma<<<dim3(M / 128, D / 64), 256, 0, stream>>>(
        obf, Wt, bo + (size_t)l * D, tmp, nullptr, M, D, D, 0);
    residual_ln<<<M, 256, 0, stream>>>(x, tmp, g1 + (size_t)l * D,
                                       be1 + (size_t)l * D, xb);
    // --- FFN block ---
    wtranspose<<<(D * DFF + 255) / 256, 256, 0, stream>>>(
        W1 + (size_t)l * D * DFF, Wt, D, DFF);
    gemm_bf16_wmma<<<dim3(M / 128, DFF / 64), 256, 0, stream>>>(
        xb, Wt, bf1 + (size_t)l * DFF, nullptr, hbf, M, DFF, D, 1);
    wtranspose<<<(DFF * D + 255) / 256, 256, 0, stream>>>(
        W2 + (size_t)l * DFF * D, Wt, DFF, D);
    gemm_bf16_wmma<<<dim3(M / 128, D / 64), 256, 0, stream>>>(
        hbf, Wt, bf2 + (size_t)l * D, tmp, nullptr, M, D, DFF, 0);
    residual_ln<<<M, 256, 0, stream>>>(x, tmp, g2 + (size_t)l * D,
                                       be2 + (size_t)l * D, xb);
  }
}
